// PointCloudTokenizer_81192061764451
// MI455X (gfx1250) — matrix operand order repaired and checked by reading, not compiled, not run on hardware
//
#include <hip/hip_runtime.h>
#include <hip/hip_bf16.h>

#define Bsz  32
#define Npts 8192
#define KCL  64
#define C1   128
#define C2   256
#define CF   512
#define OUTC 384
#define MT   128   // points per workgroup tile

typedef __attribute__((ext_vector_type(16))) __bf16 v16bf;
typedef __attribute__((ext_vector_type(8)))  float  v8f;

union BFrag { uint4 q[2]; v16bf v; };

__device__ __forceinline__ unsigned short f2bf(float x) {
  unsigned int u = __float_as_uint(x);
  unsigned int r = (u + 0x7FFFu + ((u >> 16) & 1u)) >> 16;   // RNE
  return (unsigned short)r;
}

__device__ __forceinline__ unsigned int f2bf_pk(float lo, float hi) {
  return (unsigned int)f2bf(lo) | ((unsigned int)f2bf(hi) << 16);
}

// Native CDNA5 float atomic max (no return, device scope). Memory inited to -inf.
__device__ __forceinline__ void atomicMaxF32(float* addr, float val) {
  asm volatile("global_atomic_max_num_f32 %0, %1, off scope:SCOPE_DEV"
               :: "v"(addr), "v"(val) : "memory");
}

// A fragment (16x32 bf16, M x K) from LDS, row-major with given stride (ushorts)
__device__ __forceinline__ v16bf load_a_frag(const unsigned short* lds, int row,
                                             int stride, int kbase, int lane) {
  int m   = lane & 15;
  int hi8 = (lane >> 4) << 3;                 // lanes>=16 hold K offset +8
  const unsigned short* rp = lds + (row + m) * stride + kbase + hi8;
  BFrag f;
  f.q[0] = *(const uint4*)(rp);               // K = kbase+hi8    .. +7
  f.q[1] = *(const uint4*)(rp + 16);          // K = kbase+hi8+16 .. +23
  return f.v;
}

// B fragment (32x16 bf16, K x N) where B[k][n] = W[n][k], W row-major [O][Kdim] bf16 in global
__device__ __forceinline__ v16bf load_b_frag(const unsigned short* w, int o,
                                             int Kdim, int kbase, int lane) {
  int n   = lane & 15;
  int sel = (lane >> 4) << 3;                 // lanes>=16 hold K offset +8
  const unsigned short* rp = w + (size_t)(o + n) * Kdim + kbase + sel;
  BFrag f;
  f.q[0] = *(const uint4*)(rp);               // k = kbase+sel    .. +7
  f.q[1] = *(const uint4*)(rp + 16);          // k = kbase+sel+16 .. +23
  return f.v;
}

__device__ __forceinline__ v8f bfwmma(v16bf a, v16bf b, v8f c) {
  return __builtin_amdgcn_wmma_f32_16x16x32_bf16(false, a, false, b,
                                                 (short)0, c, false, false);
}

// ---------------------------------------------------------------- utilities
__global__ void fill_u32_kernel(unsigned int* p, unsigned int val, int n) {
  int i = blockIdx.x * blockDim.x + threadIdx.x;
  if (i < n) p[i] = val;
}

__global__ void cvt_bf16_kernel(const float* src, unsigned short* dst, int n) {
  int i = blockIdx.x * blockDim.x + threadIdx.x;
  if (i < n) dst[i] = f2bf(src[i]);
}

// ---------------------------------------------------------------- phase 1
// h = relu(bn1(xyz@w1^T + b1)) -> LDS bf16 [MT][128]
// feat = h@w2^T + b2 (WMMA)    -> featWs bf16 [B*N][256], atomic-max into fg f32 [B][64][256]
__global__ __launch_bounds__(256) void phase1_kernel(
    const float* __restrict__ xyz, const int* __restrict__ choice,
    const float* __restrict__ w1, const float* __restrict__ b1,
    const float* __restrict__ g1, const float* __restrict__ bb1,
    const float* __restrict__ m1, const float* __restrict__ v1,
    const unsigned short* __restrict__ w2bf, const float* __restrict__ b2,
    unsigned short* __restrict__ featWs, float* __restrict__ fg)
{
  __shared__ __align__(16) unsigned short hTile[MT * C1];
  __shared__ int cIdx[MT];
  const int tid   = threadIdx.x;
  const int pbase = blockIdx.x * MT;
  const int bIdx  = pbase / Npts;           // MT divides Npts -> uniform per block

  { // stage A: per-point MLP1 (K=3) on VALU, 2 threads per point
    int r = tid >> 1, half = tid & 1;
    int p = pbase + r;
    float x = xyz[(size_t)p * 3 + 0];
    float y = xyz[(size_t)p * 3 + 1];
    float z = xyz[(size_t)p * 3 + 2];
    if (half == 0) cIdx[r] = choice[p];
    int c0 = half * 64;
    for (int c = c0; c < c0 + 64; ++c) {
      float s = g1[c] * rsqrtf(v1[c] + 1e-5f);
      float t = bb1[c] - m1[c] * s;
      float h = fmaf(w1[c * 3 + 0], x,
                fmaf(w1[c * 3 + 1], y,
                fmaf(w1[c * 3 + 2], z, b1[c])));
      h = fmaf(h, s, t);
      hTile[r * C1 + c] = f2bf(h > 0.f ? h : 0.f);
    }
  }
  __syncthreads();

  // stage B: feat = h @ w2^T  (M=16/wave, K=128, N=256)
  const int lane = tid & 31, wave = tid >> 5;
  const int wrow = wave * 16;
  const int o_n  = lane & 15;
  const int mhi  = (lane >> 4) * 8;

  int cl[8];
#pragma unroll
  for (int v = 0; v < 8; ++v) cl[v] = cIdx[wrow + v + mhi];

  for (int nt = 0; nt < C2 / 16; ++nt) {
    v8f acc = {};
#pragma unroll
    for (int ks = 0; ks < C1 / 32; ++ks) {
      v16bf a = load_a_frag(hTile, wrow, C1, ks * 32, lane);
      v16bf b = load_b_frag(w2bf, nt * 16, C1, ks * 32, lane);
      acc = bfwmma(a, b, acc);
    }
    int o = nt * 16 + o_n;
    float bias = b2[o];
#pragma unroll
    for (int v = 0; v < 8; ++v) {
      float fv = acc[v] + bias;
      int row = wrow + v + mhi;
      featWs[(size_t)(pbase + row) * C2 + o] = f2bf(fv);
      atomicMaxF32(&fg[((size_t)bIdx * KCL + cl[v]) * C2 + o], fv);
    }
  }
}

// ---------------------------------------------------------------- phase 2
// f = [fg[choice[p]] , feat[p]] (bf16, LDS [MT][512])
// h2 = relu(bn2(f@w3^T + b3)) -> LDS bf16 ; out = h2@w4^T + b4 -> atomic-max d_out
// Wave decomposition: 8 waves = 4 row-groups (32 rows each) x 2 N-halves.
// Each B fragment feeds two WMMAs (two A tiles) -> weight traffic from L2 halved.
__global__ __launch_bounds__(256) void phase2_kernel(
    const int* __restrict__ choice,
    const unsigned short* __restrict__ featWs, const float* __restrict__ fg,
    const unsigned short* __restrict__ w3bf, const float* __restrict__ b3,
    const float* __restrict__ g2, const float* __restrict__ bb2,
    const float* __restrict__ m2, const float* __restrict__ v2,
    const unsigned short* __restrict__ w4bf, const float* __restrict__ b4,
    float* __restrict__ outp)
{
  __shared__ __align__(16) unsigned short fTile [MT * CF];  // 128 KB
  __shared__ __align__(16) unsigned short h2Tile[MT * CF];  // 128 KB
  __shared__ int cIdx[MT];
  const int tid   = threadIdx.x;
  const int pbase = blockIdx.x * MT;
  const int bIdx  = pbase / Npts;

  { // build f tile: 2 threads per point, halves = [gathered fg | feat]
    int r = tid >> 1, half = tid & 1;
    int p = pbase + r;
    int c = choice[p];
    if (half == 0) {
      cIdx[r] = c;
      const float4* src = (const float4*)&fg[((size_t)bIdx * KCL + c) * C2];
      uint2* dst = (uint2*)&fTile[r * CF];
      for (int j = 0; j < C2 / 4; ++j) {
        float4 v = src[j];
        uint2 pk = { f2bf_pk(v.x, v.y), f2bf_pk(v.z, v.w) };
        dst[j] = pk;
      }
    } else {
      const uint4* src = (const uint4*)&featWs[(size_t)p * C2];
      uint4* dst = (uint4*)&fTile[r * CF + C2];
      for (int j = 0; j < C2 / 8; ++j) dst[j] = src[j];
    }
  }
  __syncthreads();

  const int lane   = tid & 31, wave = tid >> 5;
  const int rowg   = wave & 3;          // 4 row groups of 32 rows
  const int nthalf = wave >> 2;         // 2 N-halves
  const int wrow   = rowg * 32;
  const int o_n    = lane & 15;
  const int mhi    = (lane >> 4) * 8;

  // GEMM1: h2 = relu(bn2(f @ w3^T + b3))   M=32/wave, K=512, N=256/wave (blocked by 4)
  for (int ntb = nthalf * 16; ntb < nthalf * 16 + 16; ntb += 4) {
    v8f acc0[4] = {{}, {}, {}, {}};
    v8f acc1[4] = {{}, {}, {}, {}};
    for (int ks = 0; ks < CF / 32; ++ks) {
      v16bf a0 = load_a_frag(fTile, wrow,      CF, ks * 32, lane);
      v16bf a1 = load_a_frag(fTile, wrow + 16, CF, ks * 32, lane);
#pragma unroll
      for (int j = 0; j < 4; ++j) {
        v16bf b = load_b_frag(w3bf, (ntb + j) * 16, CF, ks * 32, lane);
        acc0[j] = bfwmma(a0, b, acc0[j]);
        acc1[j] = bfwmma(a1, b, acc1[j]);
      }
    }
#pragma unroll
    for (int j = 0; j < 4; ++j) {
      int o = (ntb + j) * 16 + o_n;
      float s = g2[o] * rsqrtf(v2[o] + 1e-5f);
      float t = bb2[o] - m2[o] * s;
      float bias = b3[o];
#pragma unroll
      for (int v = 0; v < 8; ++v) {
        float h0 = fmaf(acc0[j][v] + bias, s, t);
        float h1 = fmaf(acc1[j][v] + bias, s, t);
        h2Tile[(wrow + v + mhi) * CF + o]      = f2bf(h0 > 0.f ? h0 : 0.f);
        h2Tile[(wrow + 16 + v + mhi) * CF + o] = f2bf(h1 > 0.f ? h1 : 0.f);
      }
    }
  }
  __syncthreads();   // h2 rows are written by both N-half waves

  int cl0[8], cl1[8];
#pragma unroll
  for (int v = 0; v < 8; ++v) {
    cl0[v] = cIdx[wrow + v + mhi];
    cl1[v] = cIdx[wrow + 16 + v + mhi];
  }

  // GEMM2: out = h2 @ w4^T + b4   M=32/wave, K=512, N=192/wave (blocked by 4), scatter-max
  for (int ntb = nthalf * 12; ntb < nthalf * 12 + 12; ntb += 4) {
    v8f acc0[4] = {{}, {}, {}, {}};
    v8f acc1[4] = {{}, {}, {}, {}};
    for (int ks = 0; ks < CF / 32; ++ks) {
      v16bf a0 = load_a_frag(h2Tile, wrow,      CF, ks * 32, lane);
      v16bf a1 = load_a_frag(h2Tile, wrow + 16, CF, ks * 32, lane);
#pragma unroll
      for (int j = 0; j < 4; ++j) {
        v16bf b = load_b_frag(w4bf, (ntb + j) * 16, CF, ks * 32, lane);
        acc0[j] = bfwmma(a0, b, acc0[j]);
        acc1[j] = bfwmma(a1, b, acc1[j]);
      }
    }
#pragma unroll
    for (int j = 0; j < 4; ++j) {
      int o = (ntb + j) * 16 + o_n;
      float bias = b4[o];
      float* obase = outp + (size_t)bIdx * KCL * OUTC + o;
#pragma unroll
      for (int v = 0; v < 8; ++v) {
        atomicMaxF32(obase + (size_t)cl0[v] * OUTC, acc0[j][v] + bias);
        atomicMaxF32(obase + (size_t)cl1[v] * OUTC, acc1[j][v] + bias);
      }
    }
  }
}

// ---------------------------------------------------------------- launch
extern "C" void kernel_launch(void* const* d_in, const int* in_sizes, int n_in,
                              void* d_out, int out_size, void* d_ws, size_t ws_size,
                              hipStream_t stream) {
  const float* xyz    = (const float*)d_in[0];
  const int*   choice = (const int*)  d_in[1];
  const float* w1     = (const float*)d_in[2];
  const float* b1     = (const float*)d_in[3];
  const float* g1     = (const float*)d_in[4];
  const float* bb1    = (const float*)d_in[5];
  const float* m1     = (const float*)d_in[6];
  const float* v1     = (const float*)d_in[7];
  const float* w2     = (const float*)d_in[8];
  const float* b2     = (const float*)d_in[9];
  const float* w3     = (const float*)d_in[10];
  const float* b3     = (const float*)d_in[11];
  const float* g2     = (const float*)d_in[12];
  const float* bb2    = (const float*)d_in[13];
  const float* m2     = (const float*)d_in[14];
  const float* v2     = (const float*)d_in[15];
  const float* w4     = (const float*)d_in[16];
  const float* b4     = (const float*)d_in[17];
  float* outp = (float*)d_out;

  char* ws = (char*)d_ws;
  unsigned short* featWs = (unsigned short*)ws;
  size_t off = (size_t)Bsz * Npts * C2 * sizeof(unsigned short);   // 128 MB
  float* fg = (float*)(ws + off);           off += (size_t)Bsz * KCL * C2 * sizeof(float);
  unsigned short* w2bf = (unsigned short*)(ws + off); off += (size_t)C2 * C1 * 2;
  unsigned short* w3bf = (unsigned short*)(ws + off); off += (size_t)CF * CF * 2;
  unsigned short* w4bf = (unsigned short*)(ws + off); off += (size_t)OUTC * CF * 2;

  const unsigned int NEG_INF = 0xFF800000u;
  int nfg  = Bsz * KCL * C2;
  int nout = Bsz * KCL * OUTC;
  fill_u32_kernel<<<(nfg  + 255) / 256, 256, 0, stream>>>((unsigned int*)fg,   NEG_INF, nfg);
  fill_u32_kernel<<<(nout + 255) / 256, 256, 0, stream>>>((unsigned int*)outp, NEG_INF, nout);

  cvt_bf16_kernel<<<(C2 * C1  + 255) / 256, 256, 0, stream>>>(w2, w2bf, C2 * C1);
  cvt_bf16_kernel<<<(CF * CF  + 255) / 256, 256, 0, stream>>>(w3, w3bf, CF * CF);
  cvt_bf16_kernel<<<(OUTC * CF + 255) / 256, 256, 0, stream>>>(w4, w4bf, OUTC * CF);

  int nblocks = (Bsz * Npts) / MT;   // 2048
  phase1_kernel<<<nblocks, 256, 0, stream>>>(xyz, choice, w1, b1, g1, bb1, m1, v1,
                                             w2bf, b2, featWs, fg);
  phase2_kernel<<<nblocks, 256, 0, stream>>>(choice, featWs, fg, w3bf, b3,
                                             g2, bb2, m2, v2, w4bf, b4, outp);
}
